// GCNN_33483565040041
// MI455X (gfx1250) — matrix operand order repaired and checked by reading, not compiled, not run on hardware
//
#include <hip/hip_runtime.h>

#define N_NODES   100000
#define N_EDGES   1600000
#define F_IN      32
#define HID       64
#define N_GRAPHS  64
#define M_TILES   (N_NODES / 16)   // 6250, exact

typedef __attribute__((ext_vector_type(16))) __bf16 v16bf;
typedef __attribute__((ext_vector_type(8)))  float  v8f;

// float -> bf16 via compiler fptrunc: on gfx1250 this lowers to the native
// packed conversion (v_cvt_pk_bf16_f32 class) instead of a 3-op bit sequence.
__device__ __forceinline__ __bf16 f2bf(float f) {
    return (__bf16)f;
}

__device__ __forceinline__ void atomAddF32(float* p, float v) {
    // agent-scope relaxed fp32 add -> global_atomic_add_f32 on gfx1250
    __hip_atomic_fetch_add(p, v, __ATOMIC_RELAXED, __HIP_MEMORY_SCOPE_AGENT);
}

__global__ void zero_f32(float* __restrict__ p, int n) {
    int i = blockIdx.x * blockDim.x + threadIdx.x;
    if (i < n) p[i] = 0.0f;
}

// One-shot f32 -> bf16 conversion of all four weight matrices (12288 elements).
// Layout in out: [w1_rel 32x64 | w1_root 32x64 | w2_rel 64x64 | w2_root 64x64]
__global__ void prep_weights(const float* __restrict__ w1r, const float* __restrict__ w1o,
                             const float* __restrict__ w2r, const float* __restrict__ w2o,
                             __bf16* __restrict__ out) {
    int t = blockIdx.x * blockDim.x + threadIdx.x;
    if (t >= 12288) return;
    float v;
    if      (t < 2048) v = w1r[t];
    else if (t < 4096) v = w1o[t - 2048];
    else if (t < 8192) v = w2r[t - 4096];
    else               v = w2o[t - 8192];
    out[t] = f2bf(v);
}

// One lane per (edge, feature): coalesced gather of feat[src], fp32 atomic scatter to aggr[dst].
template <int F>
__global__ void scatter_edges(const float* __restrict__ feat, const float* __restrict__ ew,
                              const int* __restrict__ src, const int* __restrict__ dst,
                              float* __restrict__ aggr) {
    unsigned long long t = (unsigned long long)blockIdx.x * blockDim.x + threadIdx.x;
    int e = (int)(t / F);
    int f = (int)(t % F);
    if (e >= N_EDGES) return;
    int s = src[e];
    int d = dst[e];
    float v = feat[(size_t)s * F + f] * ew[e];
    atomAddF32(&aggr[(size_t)d * F + f], v);
}

// Fused dense layer: Out[n,:] = relu(A1[n,:] @ Wrel + A2[n,:] @ Wroot + bias)
// K = 32*KCHUNKS, 64 outputs. One wave per 16-node M-tile; 4 N-tiles of 16.
// Weights arrive pre-converted to bf16 (row-major [K][64]); B-tiles are gathered
// element by element, stride HID per k. Uses v_wmma_f32_16x16x32_bf16, f32 accumulation.
template <int KCHUNKS>
__global__ void dense_wmma(const float* __restrict__ A1, const float* __restrict__ A2,
                           const __bf16* __restrict__ Wrel, const __bf16* __restrict__ Wroot,
                           const float* __restrict__ bias, float* __restrict__ Out,
                           int nTiles) {
    int wave = (blockIdx.x * blockDim.x + threadIdx.x) >> 5;
    if (wave >= nTiles) return;               // wave-uniform: EXEC stays all-1 for WMMA
    const int lane  = threadIdx.x & 31;
    const int half  = lane >> 4;              // 0: lanes 0-15, 1: lanes 16-31
    const int mloc  = lane & 15;
    const int kbase = half * 8;
    const int K     = 32 * KCHUNKS;
    const int node  = wave * 16 + mloc;

    // A-matrix (16x32 bf16) VGPR layout: lane half selects K base 0/8;
    // vector elems 0-7 -> K = kbase+0..7, elems 8-15 -> K = kbase+16..23.
    v16bf a1[KCHUNKS], a2[KCHUNKS];
    const float* a1row = A1 + (size_t)node * K;
    const float* a2row = A2 + (size_t)node * K;
#pragma unroll
    for (int c = 0; c < KCHUNKS; ++c) {
#pragma unroll
        for (int i = 0; i < 16; ++i) {
            int k = c * 32 + kbase + (i < 8 ? i : 8 + i);
            a1[c][i] = f2bf(a1row[k]);
            a2[c][i] = f2bf(a2row[k]);
        }
    }

#pragma unroll
    for (int nt = 0; nt < 4; ++nt) {
        v8f acc = {};
#pragma unroll
        for (int c = 0; c < KCHUNKS; ++c) {
            // B-matrix (32x16 bf16): lane&15 = N col; elems 0-7 -> K = kbase+0..7, elems 8-15 -> K = kbase+16..23.
            v16bf brel, broot;
#pragma unroll
            for (int i = 0; i < 16; ++i) {
                int k = c * 32 + kbase + (i < 8 ? i : 8 + i);
                brel[i]  = Wrel [(size_t)k * HID + nt * 16 + mloc];
                broot[i] = Wroot[(size_t)k * HID + nt * 16 + mloc];
            }
            acc = __builtin_amdgcn_wmma_f32_16x16x32_bf16(false, a1[c], false, brel,
                                                          (short)0, acc, false, false);
            acc = __builtin_amdgcn_wmma_f32_16x16x32_bf16(false, a2[c], false, broot,
                                                          (short)0, acc, false, false);
        }
        // C/D layout: VGPR i -> M = i + 8*half, N = lane&15 (within this 16-col tile)
#pragma unroll
        for (int i = 0; i < 8; ++i) {
            int m = i + half * 8;
            int n = nt * 16 + mloc;
            float v = acc[i] + bias[n];
            v = v > 0.0f ? v : 0.0f;
            Out[(size_t)(wave * 16 + m) * HID + n] = v;
        }
    }
}

// Per-graph sum pool: one lane per (node, feature); counts from f==0 lanes.
__global__ void pool_kernel(const float* __restrict__ h, const int* __restrict__ batch,
                            float* __restrict__ sums, float* __restrict__ cnts) {
    unsigned long long t = (unsigned long long)blockIdx.x * blockDim.x + threadIdx.x;
    int n = (int)(t >> 6);
    int f = (int)(t & 63);
    if (n >= N_NODES) return;
    int g = batch[n];
    atomAddF32(&sums[(size_t)g * HID + f], h[(size_t)n * HID + f]);
    if (f == 0) atomAddF32(&cnts[g], 1.0f);
}

// Head: pooled = sums/max(cnt,1); out = relu(pooled@lw1+lb1)@lw2 + lb2. One thread per graph.
__global__ void mlp_head(const float* __restrict__ sums, const float* __restrict__ cnts,
                         const float* __restrict__ lw1, const float* __restrict__ lb1,
                         const float* __restrict__ lw2, const float* __restrict__ lb2,
                         float* __restrict__ out) {
    int g = threadIdx.x;
    if (g >= N_GRAPHS) return;
    float c = cnts[g];
    float inv = 1.0f / (c > 1.0f ? c : 1.0f);
    float acc = lb2[0];
#pragma unroll 4
    for (int j = 0; j < 16; ++j) {
        float hj = lb1[j];
        for (int f = 0; f < HID; ++f)
            hj = fmaf(sums[(size_t)g * HID + f] * inv, lw1[(size_t)f * 16 + j], hj);
        hj = hj > 0.0f ? hj : 0.0f;
        acc = fmaf(hj, lw2[j], acc);
    }
    out[g] = acc;
}

extern "C" void kernel_launch(void* const* d_in, const int* in_sizes, int n_in,
                              void* d_out, int out_size, void* d_ws, size_t ws_size,
                              hipStream_t stream) {
    const float* x       = (const float*)d_in[0];
    const float* ew      = (const float*)d_in[1];
    const float* w1_rel  = (const float*)d_in[2];
    const float* b1      = (const float*)d_in[3];
    const float* w1_root = (const float*)d_in[4];
    const float* w2_rel  = (const float*)d_in[5];
    const float* b2      = (const float*)d_in[6];
    const float* w2_root = (const float*)d_in[7];
    const float* lw1     = (const float*)d_in[8];
    const float* lb1     = (const float*)d_in[9];
    const float* lw2     = (const float*)d_in[10];
    const float* lb2     = (const float*)d_in[11];
    const int*   ei      = (const int*)d_in[12];
    const int*   batch   = (const int*)d_in[13];
    const int*   src     = ei;
    const int*   dst     = ei + N_EDGES;
    float*       out     = (float*)d_out;

    // Workspace layout (floats): bufA (aggr1 then h2), bufB (h1), bufC (aggr2),
    // sums (64x64), cnts (64), then bf16 weight cache (12288 bf16 = 24KB, 32B aligned).
    float*  ws    = (float*)d_ws;
    float*  bufA  = ws;
    float*  bufB  = ws + (size_t)N_NODES * HID;
    float*  bufC  = ws + (size_t)2 * N_NODES * HID;
    float*  sums  = ws + (size_t)3 * N_NODES * HID;
    float*  cnts  = sums + N_GRAPHS * HID;
    __bf16* wbf   = (__bf16*)(cnts + N_GRAPHS);
    __bf16* w1r_bf = wbf;
    __bf16* w1o_bf = wbf + 2048;
    __bf16* w2r_bf = wbf + 4096;
    __bf16* w2o_bf = wbf + 8192;

    auto cdiv = [](long long a, long long b) { return (int)((a + b - 1) / b); };

    // Zero accumulators + convert weights (graph-capture-safe: plain kernels on stream)
    zero_f32<<<cdiv((long long)N_NODES * F_IN, 256), 256, 0, stream>>>(bufA, N_NODES * F_IN);
    zero_f32<<<cdiv((long long)N_NODES * HID, 256), 256, 0, stream>>>(bufC, N_NODES * HID);
    zero_f32<<<cdiv(N_GRAPHS * HID + N_GRAPHS, 256), 256, 0, stream>>>(sums, N_GRAPHS * HID + N_GRAPHS);
    prep_weights<<<cdiv(12288, 256), 256, 0, stream>>>(w1_rel, w1_root, w2_rel, w2_root, wbf);

    // Layer 1: scatter (F=32) then fused WMMA dense + bias + relu -> h1 (bufB)
    scatter_edges<F_IN><<<cdiv((long long)N_EDGES * F_IN, 256), 256, 0, stream>>>(x, ew, src, dst, bufA);
    dense_wmma<1><<<cdiv((long long)M_TILES * 32, 128), 128, 0, stream>>>(
        bufA, x, w1r_bf, w1o_bf, b1, bufB, M_TILES);

    // Layer 2: scatter (F=64) then fused WMMA dense + bias + relu -> h2 (bufA reused)
    scatter_edges<HID><<<cdiv((long long)N_EDGES * HID, 256), 256, 0, stream>>>(bufB, ew, src, dst, bufC);
    dense_wmma<2><<<cdiv((long long)M_TILES * 32, 128), 128, 0, stream>>>(
        bufC, bufB, w2r_bf, w2o_bf, b2, bufA, M_TILES);

    // Mean pool + MLP head
    pool_kernel<<<cdiv((long long)N_NODES * HID, 256), 256, 0, stream>>>(bufA, batch, sums, cnts);
    mlp_head<<<1, 64, 0, stream>>>(sums, cnts, lw1, lb1, lw2, lb2, out);
}
